// _AttentionBlock_5935644803163
// MI455X (gfx1250) — matrix operand order repaired and checked by reading, not compiled, not run on hardware
//
#include <hip/hip_runtime.h>
#include <hip/hip_bf16.h>

#define B_ 16
#define C_ 512
#define T_ 1024
#define H_ 8
#define D_ 64
#define E_ 512

typedef _Float16 half_t;
typedef __attribute__((ext_vector_type(16))) _Float16 v16h;
typedef __attribute__((ext_vector_type(8)))  _Float16 v8h;
typedef __attribute__((ext_vector_type(8)))  float    v8f;
typedef __attribute__((ext_vector_type(4)))  uint32_t u32x4;
typedef __attribute__((ext_vector_type(8)))  uint32_t u32x8;

// load a 32-byte A/B fragment as two contiguous 16B chunks
__device__ __forceinline__ v16h ldg2(const half_t* p0, const half_t* p1) {
  v16h r;
  reinterpret_cast<v8h*>(&r)[0] = *reinterpret_cast<const v8h*>(p0);
  reinterpret_cast<v8h*>(&r)[1] = *reinterpret_cast<const v8h*>(p1);
  return r;
}

__device__ __forceinline__ v8f wmma16(v16h a, v16h b, v8f c) {
  return __builtin_amdgcn_wmma_f32_16x16x32_f16(false, a, false, b, (short)0, c,
                                                false, false);
}

// TDM: 2D tile load Global->LDS with per-row padding (D# per ISA 08 §8.3/8.4).
// tile = tdim0 (contig, elements) x tdim1 rows, element = 2 bytes,
// row stride in memory = stride0 elements; LDS gets +16B pad per 1KB row.
__device__ __forceinline__ void tdm_load_slab(uint32_t lds_off, const void* gp) {
  uint64_t ga = (uint64_t)(uintptr_t)gp;
  u32x4 g0;
  g0[0] = 1u;                                   // count=1, user desc
  g0[1] = lds_off;                              // lds_addr (bytes)
  g0[2] = (uint32_t)ga;                         // global_addr[31:0]
  g0[3] = (uint32_t)((ga >> 32) & 0x1ffffffu) | 0x80000000u; // ga[56:32]|type=2
  u32x8 g1;
  g1[0] = (1u << 16)      // data_size = 1 -> 2 bytes
        | (1u << 20)      // pad_enable
        | (7u << 22)      // pad_interval = 256 DWORDs (one 1KB row)
        | (3u << 25);     // pad_amount   = 4 DWORDs (16B)
  g1[1] = 512u << 16;     // tensor_dim0 = 512 (lo16 @ bits 63:48)
  g1[2] = 64u << 16;      // tensor_dim0 hi=0 | tensor_dim1 = 64 (lo16)
  g1[3] = 512u << 16;     // tensor_dim1 hi=0 | tile_dim0 = 512
  g1[4] = 64u;            // tile_dim1 = 64, tile_dim2 = 0
  g1[5] = 512u;           // tensor_dim0_stride lo32 = 512 elements
  g1[6] = 0u;
  g1[7] = 0u;
  asm volatile("tensor_load_to_lds %0, %1" :: "s"(g0), "s"(g1) : "memory");
  __builtin_amdgcn_s_wait_tensorcnt(0);
}

// ---------------- weight convert + transpose: WT[c*R+r] = (f16)W[r*Cc+c] ----
__global__ void k_wcvt(const float* __restrict__ W, half_t* __restrict__ WT,
                       int R, int Cc) {
  int i = blockIdx.x * 256 + threadIdx.x;
  if (i >= R * Cc) return;
  int c = i / R, r = i % R;
  WT[i] = (half_t)W[(size_t)r * Cc + c];
}

// ---------------- LayerNorm over C for each (b,t); writes xn f16 (B*T, C) ---
__global__ void k_ln(const float* __restrict__ X, const float* __restrict__ g,
                     const float* __restrict__ beta, half_t* __restrict__ xn) {
  int idx = blockIdx.x * blockDim.x + threadIdx.x;   // b*T + t
  int b = idx / T_, t = idx % T_;
  const float* xp = X + (size_t)b * C_ * T_ + t;
  float s = 0.f, ss = 0.f;
  for (int c = 0; c < C_; ++c) {
    float v = xp[(size_t)c * T_];
    s += v; ss += v * v;
  }
  float mean = s * (1.f / C_);
  float var  = ss * (1.f / C_) - mean * mean;
  float rstd = rsqrtf(var + 1e-6f);
  half_t* op = xn + (size_t)idx * C_;
  for (int c = 0; c < C_; ++c) {
    float v = xp[(size_t)c * T_];
    op[c] = (half_t)(((v - mean) * rstd) * g[c] + beta[c]);
  }
}

// ---------------- fused QKV GEMM, block tile 128(M) x 64(N) -----------------
// Weight slab (64 rows x 512) staged in LDS by the Tensor Data Mover with
// 16B/row padding (stride 520 halfs) to spread B-fragment ds_load_b128
// across all 64 LDS banks. Each wave: 16x64 tile, A fragment reused 4x.
__global__ void __launch_bounds__(256) k_qkv(
    const half_t* __restrict__ xn, const half_t* __restrict__ WqT,
    const half_t* __restrict__ WkT, const half_t* __restrict__ WvT,
    const float* __restrict__ bq, const float* __restrict__ bk,
    const float* __restrict__ bv, half_t* __restrict__ Q,
    half_t* __restrict__ K, half_t* __restrict__ Vt) {
  const int KS = 520;                              // padded row stride (halfs)
  __shared__ __align__(128) half_t Bs[64 * KS];    // 65 KB slab

  const int MB = (B_ * T_) / 128;                  // 128
  const int NS = E_ / 64;                          // 8
  int bx  = blockIdx.x;
  int mat = bx / (MB * NS);                        // 0=Q 1=K 2=V
  int rem = bx % (MB * NS);
  int mb = rem / NS, slab = rem % NS;
  const half_t* WT  = (mat == 0) ? WqT : (mat == 1) ? WkT : WvT;
  const float* bias = (mat == 0) ? bq  : (mat == 1) ? bk  : bv;

  if (threadIdx.x < 32) {   // one wave issues the TDM, waits TENSORcnt
    tdm_load_slab((uint32_t)(uintptr_t)(void*)&Bs[0],
                  WT + (size_t)(slab * 64) * C_);
  }
  __syncthreads();

  int wid = threadIdx.x >> 5, lane = threadIdx.x & 31;
  int l15  = lane & 15;
  int aoff = (lane < 16) ? 0 : 16;
  int boff = (lane < 16) ? 0 : 16;
  const half_t* ap = xn + (size_t)(mb * 128 + wid * 16 + l15) * C_;
  const half_t* b0 = &Bs[(size_t)(l15)      * KS];
  const half_t* b1 = &Bs[(size_t)(16 + l15) * KS];
  const half_t* b2 = &Bs[(size_t)(32 + l15) * KS];
  const half_t* b3 = &Bs[(size_t)(48 + l15) * KS];

  v8f acc0 = {}, acc1 = {}, acc2 = {}, acc3 = {};
  for (int k0 = 0; k0 < C_; k0 += 32) {
    v16h a = ldg2(ap + k0 + aoff, ap + k0 + aoff + 8);
    acc0 = wmma16(a, ldg2(b0 + k0 + boff, b0 + k0 + boff + 8), acc0);
    acc1 = wmma16(a, ldg2(b1 + k0 + boff, b1 + k0 + boff + 8), acc1);
    acc2 = wmma16(a, ldg2(b2 + k0 + boff, b2 + k0 + boff + 8), acc2);
    acc3 = wmma16(a, ldg2(b3 + k0 + boff, b3 + k0 + boff + 8), acc3);
  }

  float scale = (mat == 0) ? 0.125f : 1.0f;        // fold 1/sqrt(D) into Q
  int mbase = mb * 128 + wid * 16 + ((lane < 16) ? 0 : 8);
  v8f acc[4] = {acc0, acc1, acc2, acc3};
#pragma unroll
  for (int j = 0; j < 4; ++j) {
    int e = slab * 64 + j * 16 + l15;
    int h = e >> 6, d = e & 63;
    float bb = bias[e];
    for (int r = 0; r < 8; ++r) {
      float v = (acc[j][r] + bb) * scale;
      int m = mbase + r;
      int b_ = m >> 10, t = m & (T_ - 1);
      if (mat == 2)     // V stored transposed: (B,H,D,T)
        Vt[(((size_t)b_ * H_ + h) * D_ + d) * T_ + t] = (half_t)v;
      else {            // Q,K stored (B,H,T,D)
        half_t* dst = (mat == 0) ? Q : K;
        dst[(((size_t)b_ * H_ + h) * T_ + t) * D_ + d] = (half_t)v;
      }
    }
  }
}

// ---------------- flash attention: one wave -> 16 query rows of one head ----
__global__ void __launch_bounds__(128) k_attn(
    const half_t* __restrict__ Q, const half_t* __restrict__ K,
    const half_t* __restrict__ Vt, half_t* __restrict__ O) {
  __shared__ half_t Pbuf[4][16 * 32];   // per-wave P transpose bounce
  int wid = threadIdx.x >> 5, lane = threadIdx.x & 31;
  int wave = blockIdx.x * 4 + wid;      // B*H*(T/16) = 8192 waves
  int qt = wave & 63;
  int bh = wave >> 6;                   // b*H + h
  int h = bh & 7, b = bh >> 3;

  const half_t* Qp = Q  + ((size_t)bh * T_ + qt * 16) * D_;
  const half_t* Kp = K  + (size_t)bh * T_ * D_;
  const half_t* Vp = Vt + (size_t)bh * D_ * T_;

  int l15  = lane & 15;
  int aoff = (lane < 16) ? 0 : 16;
  int bo   = (lane < 16) ? 0 : 16;

  const half_t* qrow = Qp + (size_t)l15 * D_;
  v16h qa0 = ldg2(qrow + aoff,      qrow + aoff + 8);
  v16h qa1 = ldg2(qrow + 32 + aoff, qrow + 32 + aoff + 8);

  float mrow[8], lrow[8];
  v8f o0 = {}, o1 = {}, o2 = {}, o3 = {};
  for (int r = 0; r < 8; ++r) { mrow[r] = -1e30f; lrow[r] = 0.f; }
  half_t* pb = &Pbuf[wid][0];

  for (int kt = 0; kt < T_ / 32; ++kt) {
    int key0 = kt * 32;
    if (kt + 1 < T_ / 32) {
      __builtin_prefetch(Kp + (size_t)(key0 + 32) * D_, 0, 0);
      __builtin_prefetch(Vp + key0 + 32, 0, 0);
    }
    const half_t* kc0 = Kp + (size_t)(key0 + l15) * D_;
    const half_t* kc1 = Kp + (size_t)(key0 + 16 + l15) * D_;
    v16h kb00 = ldg2(kc0 + bo,      kc0 + bo + 8);
    v16h kb01 = ldg2(kc0 + 32 + bo, kc0 + 32 + bo + 8);
    v16h kb10 = ldg2(kc1 + bo,      kc1 + bo + 8);
    v16h kb11 = ldg2(kc1 + 32 + bo, kc1 + 32 + bo + 8);
    v8f s0 = {}, s1 = {};
    s0 = wmma16(qa0, kb00, s0); s0 = wmma16(qa1, kb01, s0);
    s1 = wmma16(qa0, kb10, s1); s1 = wmma16(qa1, kb11, s1);

    for (int r = 0; r < 8; ++r) {
      float a = s0[r], c = s1[r];
      float rm = fmaxf(a, c);
      for (int m = 1; m < 16; m <<= 1) rm = fmaxf(rm, __shfl_xor(rm, m, 32));
      float nm = fmaxf(mrow[r], rm);
      float alpha = __expf(mrow[r] - nm);
      float p0 = __expf(a - nm), p1 = __expf(c - nm);
      float rs = p0 + p1;
      for (int m = 1; m < 16; m <<= 1) rs += __shfl_xor(rs, m, 32);
      lrow[r] = lrow[r] * alpha + rs;
      mrow[r] = nm;
      o0[r] *= alpha; o1[r] *= alpha; o2[r] *= alpha; o3[r] *= alpha;
      int rowm = r + ((lane < 16) ? 0 : 8);
      pb[rowm * 32 + l15]      = (half_t)p0;
      pb[rowm * 32 + 16 + l15] = (half_t)p1;
    }
    asm volatile("s_wait_dscnt 0x0" ::: "memory");  // same-wave LDS RAW

    const half_t* prow = pb + l15 * 32;
    v16h pa;
    reinterpret_cast<v8h*>(&pa)[0] = *reinterpret_cast<const v8h*>(prow + aoff);
    reinterpret_cast<v8h*>(&pa)[1] = *reinterpret_cast<const v8h*>(prow + aoff + 8);
    const half_t* vb = Vp + key0 + ((lane < 16) ? 0 : 16);
    const half_t* v0p = vb + (size_t)(l15)      * T_;
    const half_t* v1p = vb + (size_t)(16 + l15) * T_;
    const half_t* v2p = vb + (size_t)(32 + l15) * T_;
    const half_t* v3p = vb + (size_t)(48 + l15) * T_;
    o0 = wmma16(pa, ldg2(v0p, v0p + 8), o0);
    o1 = wmma16(pa, ldg2(v1p, v1p + 8), o1);
    o2 = wmma16(pa, ldg2(v2p, v2p + 8), o2);
    o3 = wmma16(pa, ldg2(v3p, v3p + 8), o3);
  }

  for (int r = 0; r < 8; ++r) {
    float inv = 1.0f / lrow[r];
    int t = qt * 16 + r + ((lane < 16) ? 0 : 8);
    half_t* orow = O + ((size_t)b * T_ + t) * E_ + h * D_;
    orow[l15]      = (half_t)(o0[r] * inv);
    orow[16 + l15] = (half_t)(o1[r] * inv);
    orow[32 + l15] = (half_t)(o2[r] * inv);
    orow[48 + l15] = (half_t)(o3[r] * inv);
  }
}

// ------ output projection + bias + residual + transpose; 16x64 per wave ----
__global__ void __launch_bounds__(256) k_oproj(
    const half_t* __restrict__ O, const half_t* __restrict__ WoT,
    const float* __restrict__ bo, const float* __restrict__ X,
    float* __restrict__ out) {
  const int NS = C_ / 64;          // 8 column slabs
  int wave = (blockIdx.x * blockDim.x + threadIdx.x) >> 5;
  int lane = threadIdx.x & 31;
  int mt = wave / NS, ns = wave % NS;
  int l15  = lane & 15;
  int aoff = (lane < 16) ? 0 : 16;
  int boff = (lane < 16) ? 0 : 16;
  const half_t* ap = O + (size_t)(mt * 16 + l15) * E_;
  const half_t* b0 = WoT + (size_t)(ns * 64 + l15)      * E_;
  const half_t* b1 = WoT + (size_t)(ns * 64 + 16 + l15) * E_;
  const half_t* b2 = WoT + (size_t)(ns * 64 + 32 + l15) * E_;
  const half_t* b3 = WoT + (size_t)(ns * 64 + 48 + l15) * E_;

  v8f acc0 = {}, acc1 = {}, acc2 = {}, acc3 = {};
  for (int k0 = 0; k0 < E_; k0 += 32) {
    v16h a = ldg2(ap + k0 + aoff, ap + k0 + aoff + 8);
    acc0 = wmma16(a, ldg2(b0 + k0 + boff, b0 + k0 + boff + 8), acc0);
    acc1 = wmma16(a, ldg2(b1 + k0 + boff, b1 + k0 + boff + 8), acc1);
    acc2 = wmma16(a, ldg2(b2 + k0 + boff, b2 + k0 + boff + 8), acc2);
    acc3 = wmma16(a, ldg2(b3 + k0 + boff, b3 + k0 + boff + 8), acc3);
  }

  int mbase = mt * 16 + ((lane < 16) ? 0 : 8);
  v8f acc[4] = {acc0, acc1, acc2, acc3};
#pragma unroll
  for (int j = 0; j < 4; ++j) {
    int c = ns * 64 + j * 16 + l15;
    float bb = bo[c];
    for (int r = 0; r < 8; ++r) {
      int m = mbase + r;
      int b_ = m >> 10, t = m & (T_ - 1);
      size_t xi = ((size_t)b_ * C_ + c) * T_ + t;
      out[xi] = X[xi] + acc[j][r] + bb;
    }
  }
}

extern "C" void kernel_launch(void* const* d_in, const int* in_sizes, int n_in,
                              void* d_out, int out_size, void* d_ws, size_t ws_size,
                              hipStream_t stream) {
  const float* X    = (const float*)d_in[0];
  const float* ln_g = (const float*)d_in[1];
  const float* ln_b = (const float*)d_in[2];
  const float* Wq   = (const float*)d_in[3];
  const float* bq   = (const float*)d_in[4];
  const float* Wk   = (const float*)d_in[5];
  const float* bk   = (const float*)d_in[6];
  const float* Wv   = (const float*)d_in[7];
  const float* bv   = (const float*)d_in[8];
  const float* Wo   = (const float*)d_in[9];
  const float* bo   = (const float*)d_in[10];
  float* out = (float*)d_out;

  const size_t WSZ = (size_t)C_ * E_;          // 262144
  const size_t ASZ = (size_t)B_ * T_ * C_;     // 8388608
  half_t* ws  = (half_t*)d_ws;
  half_t* WqT = ws;
  half_t* WkT = WqT + WSZ;
  half_t* WvT = WkT + WSZ;
  half_t* WoT = WvT + WSZ;
  half_t* xn  = WoT + WSZ;
  half_t* Qs  = xn  + ASZ;
  half_t* Ks  = Qs  + ASZ;
  half_t* Vt  = Ks  + ASZ;
  half_t* Of  = Vt  + ASZ;

  // 1. weights -> f16 transposed
  int wblk = (int)((WSZ + 255) / 256);
  k_wcvt<<<wblk, 256, 0, stream>>>(Wq, WqT, C_, E_);
  k_wcvt<<<wblk, 256, 0, stream>>>(Wk, WkT, C_, E_);
  k_wcvt<<<wblk, 256, 0, stream>>>(Wv, WvT, C_, E_);
  k_wcvt<<<wblk, 256, 0, stream>>>(Wo, WoT, E_, C_);

  // 2. layernorm (B*T tokens)
  k_ln<<<(B_ * T_) / 256, 256, 0, stream>>>(X, ln_g, ln_b, xn);

  // 3. fused QKV GEMM: 3 mats x 128 m-blocks x 8 n-slabs
  k_qkv<<<3 * ((B_ * T_) / 128) * (E_ / 64), 256, 0, stream>>>(
      xn, WqT, WkT, WvT, bq, bk, bv, Qs, Ks, Vt);

  // 4. flash attention: B*H*(T/16) waves, 4 waves/block
  k_attn<<<(B_ * H_ * (T_ / 16)) / 4, 128, 0, stream>>>(Qs, Ks, Vt, Of);

  // 5. output projection + residual + transpose
  k_oproj<<<((B_ * T_ / 16) * (C_ / 64)) / 8, 256, 0, stream>>>(
      Of, WoT, bo, X, out);
}
